// AttnModel_66554813218832
// MI455X (gfx1250) — matrix operand order repaired and checked by reading, not compiled, not run on hardware
//
#include <hip/hip_runtime.h>

typedef __attribute__((ext_vector_type(16))) __bf16 v16bf;
typedef __attribute__((ext_vector_type(8)))  float  v8f;

#define D_     768
#define NN     200
#define NP     208
#define DKH    192
#define PITCH1 776                       // row-major chunk: halves/row (1552 B, 16B-aligned)
#define PITCH2 34                        // f-major chunk: halves/row (68 B, 4B-aligned, 17-dword bank stride)
#define CHUNK_REGION 52224               // max(32*776*2 = 49664, 768*34*2 = 52224)
#define S_OFF   CHUNK_REGION             // scores/attn: float[4][208] = 3328 B
#define RU_OFF  (S_OFF + 4*NP*4)         // R: bf16[4][768] = 6144 B
#define QIN_OFF (RU_OFF + 4*D_*2)        // qin: float[512] = 2048 B
#define SMEM_BYTES (QIN_OFF + 512*4)     // 63744 B total
#define UST_OFF 3072                     // U staging (phase A only), inside chunk region

union FragBF { unsigned short s[16]; unsigned int w[8]; uint4 q[2]; v16bf v; };

static __device__ inline unsigned short f2bf(float x) {
  unsigned u = __float_as_uint(x);
  u += 0x7fffu + ((u >> 16) & 1u);          // round-to-nearest-even
  return (unsigned short)(u >> 16);
}
static __device__ inline float bf2f(unsigned short h) {
  return __uint_as_float((unsigned)h << 16);
}
static __device__ inline float wred_max(float v) {
  for (int o = 16; o; o >>= 1) v = fmaxf(v, __shfl_xor(v, o, 32));
  return v;
}
static __device__ inline float wred_sum(float v) {
  for (int o = 16; o; o >>= 1) v += __shfl_xor(v, o, 32);
  return v;
}

static __device__ inline float4 load_kin4(int b, int ng, int fb,
                                          const float* seq, const float* seq_e,
                                          const float* seq_t, bool prefetch_next) {
  const float* sp; int off;
  if (fb < 256)      { sp = seq;   off = fb;       }
  else if (fb < 512) { sp = seq_e; off = fb - 256; }
  else               { sp = seq_t; off = fb - 512; }
  const float* p = sp + ((size_t)b * NN + ng) * 256 + off;
  if (prefetch_next && ng + 32 < NN)
    __builtin_prefetch(p + 32 * 256, 0, 0);        // global_prefetch_b8: next chunk
  return *(const float4*)p;
}

// Pass-1 staging: row-major bf16 chunk[n][f], pitch PITCH1.
static __device__ inline void stage_row_major(unsigned short* s_chunk, int b, int n0,
                                              const float* seq, const float* seq_e,
                                              const float* seq_t, int tid) {
  for (int v = tid; v < 32 * (D_ / 4); v += 256) {
    const int n  = v / (D_ / 4);
    const int fb = (v - n * (D_ / 4)) * 4;
    const int ng = n0 + n;
    float4 val = make_float4(0.f, 0.f, 0.f, 0.f);
    if (ng < NN) val = load_kin4(b, ng, fb, seq, seq_e, seq_t, true);
    unsigned short* dst = &s_chunk[n * PITCH1 + fb];
    dst[0] = f2bf(val.x); dst[1] = f2bf(val.y);
    dst[2] = f2bf(val.z); dst[3] = f2bf(val.w);
  }
}

// Pass-2 staging: f-major bf16 chunk_t[f][n], pitch PITCH2 (transposed in LDS).
static __device__ inline void stage_f_major(unsigned short* s_chunkT, int b, int n0,
                                            const float* seq, const float* seq_e,
                                            const float* seq_t, int tid) {
  for (int v = tid; v < 32 * (D_ / 4); v += 256) {
    const int n  = v / (D_ / 4);
    const int fb = (v - n * (D_ / 4)) * 4;
    const int ng = n0 + n;
    float4 val = make_float4(0.f, 0.f, 0.f, 0.f);
    if (ng < NN) val = load_kin4(b, ng, fb, seq, seq_e, seq_t, false);
    s_chunkT[(fb + 0) * PITCH2 + n] = f2bf(val.x);
    s_chunkT[(fb + 1) * PITCH2 + n] = f2bf(val.y);
    s_chunkT[(fb + 2) * PITCH2 + n] = f2bf(val.z);
    s_chunkT[(fb + 3) * PITCH2 + n] = f2bf(val.w);
  }
}

__global__ __launch_bounds__(256, 1)
void attn_fused(const float* __restrict__ src,   const float* __restrict__ src_t,
                const float* __restrict__ seq,   const float* __restrict__ seq_t,
                const float* __restrict__ seq_e, const unsigned char* __restrict__ mask,
                const float* __restrict__ Wq,    const float* __restrict__ Wk,
                const float* __restrict__ Wv,    const float* __restrict__ Wfc,
                const float* __restrict__ bfc,   const float* __restrict__ ln_g,
                const float* __restrict__ ln_b,  const float* __restrict__ fc1_w,
                const float* __restrict__ fc1_b, const float* __restrict__ fc2_w,
                const float* __restrict__ fc2_b, float* __restrict__ out) {
  const int b    = blockIdx.x;
  const int tid  = threadIdx.x;
  const int wv   = tid >> 5;
  const int lane = tid & 31;
  const int hi   = lane >> 4;    // half-wave index
  const int col  = lane & 15;

  __shared__ __align__(16) unsigned char smem[SMEM_BYTES];
  unsigned short* s_chunk  = (unsigned short*)smem;                 // pass-1 row-major
  unsigned short* s_chunkT = (unsigned short*)smem;                 // pass-2 f-major
  float*          s_S      = (float*)(smem + S_OFF);
  unsigned short* s_R      = (unsigned short*)(smem + RU_OFF);
  float*          s_qin    = (float*)(smem + QIN_OFF);

  // ---- Phase A1: load src / src_t (q_in middle 256 is structurally zero) ----
  for (int i = tid; i < 512; i += 256)
    s_qin[i] = (i < 256) ? src[(size_t)b * 256 + i]
                         : src_t[(size_t)b * 256 + (i - 256)];
  __syncthreads();

  // ---- Phase A2: qh = q_in @ Wq (skip zero rows 256..511) -> chunk region ----
  float* s_qh = (float*)smem;
  for (int d = tid; d < D_; d += 256) {
    float acc = 0.f;
    for (int f = 0; f < 256; ++f) acc += s_qin[f]       * Wq[(size_t)f        * D_ + d];
    for (int t = 0; t < 256; ++t) acc += s_qin[256 + t] * Wq[(size_t)(512 + t) * D_ + d];
    s_qh[d] = acc;
  }
  __syncthreads();

  // ---- Phase A3: U[h][f] = sum_dk Wk[f, h*192+dk] * qh[h*192+dk]; zero s_S ----
  unsigned short* s_Ust = (unsigned short*)(smem + UST_OFF);        // chunk region, past qh
  for (int i = tid; i < 4 * NP; i += 256) s_S[i] = 0.f;
  for (int f = tid; f < D_; f += 256) {
    const float* wrow = Wk + (size_t)f * D_;
    #pragma unroll
    for (int h = 0; h < 4; ++h) {
      float acc = 0.f;
      const float* qh_h = s_qh + h * DKH;
      const float* wk_h = wrow + h * DKH;
      for (int k = 0; k < DKH; ++k) acc += wk_h[k] * qh_h[k];
      s_Ust[h * D_ + f] = f2bf(acc);
    }
  }
  __syncthreads();

  // ---- Hoist pass-1 A fragments (U) into registers: reused across all chunks ----
  const int kt0 = (wv & 3) * 6;          // this wave's 6 k-tiles of K=768
  const int nt  = wv >> 2;               // this wave's n-tile within the chunk
  FragBF ua[6];
  #pragma unroll
  for (int k = 0; k < 6; ++k) {
    if (col < 4) {                       // A row = head (4 live rows)
      const uint4* up = (const uint4*)&s_Ust[col * D_ + (kt0 + k) * 32 + hi * 8];
      ua[k].q[0] = up[0];
      ua[k].q[1] = up[2];                // halves e>=8 sit at f+16
    } else {
      ua[k].q[0] = make_uint4(0u,0u,0u,0u); ua[k].q[1] = make_uint4(0u,0u,0u,0u);
    }
  }
  __syncthreads();                        // staging may now overwrite U

  // ---- Pass 1 (WMMA): S^T = U @ k_in^T, streamed in 7 chunks of 32 rows ----
  for (int c = 0; c < 7; ++c) {
    stage_row_major(s_chunk, b, c * 32, seq, seq_e, seq_t, tid);
    __syncthreads();
    v8f acc = {0.f,0.f,0.f,0.f,0.f,0.f,0.f,0.f};
    #pragma unroll
    for (int k = 0; k < 6; ++k) {
      // B: k_in^T tile 32x16; col = n-local, K = f -> halves f+hi*8 (e<8), f+16+hi*8 (e>=8)
      FragBF bb;
      const uint4* bp =
          (const uint4*)&s_chunk[(nt * 16 + col) * PITCH1 + (kt0 + k) * 32 + hi * 8];
      bb.q[0] = bp[0]; bb.q[1] = bp[2];
      acc = __builtin_amdgcn_wmma_f32_16x16x32_bf16(false, ua[k].v, false, bb.v,
                                                    (short)0, acc, false, false);
    }
    if (hi == 0) {                        // D rows 0..3 = heads; partial-K reduce
      const int n = c * 32 + nt * 16 + col;
      if (n < NP) {
        #pragma unroll
        for (int i = 0; i < 4; ++i) atomicAdd(&s_S[i * NP + n], acc[i]);
      }
    }
    __syncthreads();
  }

  // ---- Softmax over N per head (waves 0..3, one head each) ----
  if (wv < 4) {
    const int h = wv;
    const float rs = 0.0721687836f;      // 1/sqrt(192)
    float mx = -1e30f;
    for (int n = lane; n < NP; n += 32) {
      float v = -1e30f;
      if (n < NN) {
        v = s_S[h * NP + n] * rs;
        if (mask[(size_t)b * NN + n]) v = -1e10f;
      }
      s_S[h * NP + n] = v;
      mx = fmaxf(mx, v);
    }
    mx = wred_max(mx);
    float sum = 0.f;
    for (int n = lane; n < NP; n += 32) {
      float e = (n < NN) ? __expf(s_S[h * NP + n] - mx) : 0.f;
      s_S[h * NP + n] = e;
      sum += e;
    }
    sum = wred_sum(sum);
    const float inv = 1.f / sum;
    for (int n = lane; n < NP; n += 32) s_S[h * NP + n] *= inv;
  }
  __syncthreads();

  // ---- Pass 2 (WMMA): R[h,f] = attn @ k_in, chunk staged f-major in LDS ----
  v8f racc[6];
  #pragma unroll
  for (int t = 0; t < 6; ++t) racc[t] = (v8f){0.f,0.f,0.f,0.f,0.f,0.f,0.f,0.f};
  for (int c = 0; c < 7; ++c) {
    stage_f_major(s_chunkT, b, c * 32, seq, seq_e, seq_t, tid);
    __syncthreads();
    FragBF aa;                            // A: attn 16x32, rows = heads (4 live)
    if (col < 4) {
      const int nb = c * 32 + hi * 16;
      #pragma unroll
      for (int e = 0; e < 8; ++e) {
        const int n = nb + e;
        aa.s[e] = (n < NP) ? f2bf(s_S[col * NP + n]) : (unsigned short)0;
      }
      #pragma unroll
      for (int e = 8; e < 16; ++e) {
        const int n = nb + e;
        aa.s[e] = (n < NP) ? f2bf(s_S[col * NP + n]) : (unsigned short)0;
      }
    } else {
      aa.q[0] = make_uint4(0u,0u,0u,0u); aa.q[1] = make_uint4(0u,0u,0u,0u);
    }
    #pragma unroll
    for (int t = 0; t < 6; ++t) {
      const int ft = wv * 6 + t;          // 48 f-tiles over 8 waves
      // B: k_in tile 32x16; col = f-local, K = n -> contiguous 16 halves (8 dwords)
      FragBF bb;
      const unsigned int* bp =
          (const unsigned int*)&s_chunkT[(ft * 16 + col) * PITCH2 + hi * 16];
      #pragma unroll
      for (int j = 0; j < 8; ++j) bb.w[j] = bp[j];
      racc[t] = __builtin_amdgcn_wmma_f32_16x16x32_bf16(false, aa.v, false, bb.v,
                                                        (short)0, racc[t], false, false);
    }
    __syncthreads();
  }
  if (hi == 0) {                          // D rows 0..3 = heads
    #pragma unroll
    for (int t = 0; t < 6; ++t) {
      const int ft = wv * 6 + t;
      #pragma unroll
      for (int i = 0; i < 4; ++i)
        s_R[i * D_ + ft * 16 + col] = f2bf(racc[t][i]);
    }
  }
  __syncthreads();

  // ---- Phase E (VALU, L2-resident weights), scratch overlays chunk region ----
  float* s_ctx = (float*)smem;
  float* s_x   = s_ctx + D_;
  float* s_hdn = s_x + 1024;
  float* s_red = s_hdn + 256;

  // ctx[d] = sum_f R[h(d)][f] * Wv[f][d]
  for (int d = tid; d < D_; d += 256) {
    const unsigned short* r = s_R + (d / DKH) * D_;
    float acc = 0.f;
    for (int f = 0; f < D_; ++f) acc += bf2f(r[f]) * Wv[(size_t)f * D_ + d];
    s_ctx[d] = acc;
  }
  __syncthreads();

  // out[d] = ctx @ Wfc + bfc + q_in (residual; middle 256 is zero)
  for (int d = tid; d < D_; d += 256) {
    float acc = bfc[d];
    if (d < 256) acc += s_qin[d];
    else if (d >= 512) acc += s_qin[d - 256];
    for (int k = 0; k < D_; ++k) acc += s_ctx[k] * Wfc[(size_t)k * D_ + d];
    s_x[d] = acc;
  }
  __syncthreads();

  // LayerNorm over D
  float ps = 0.f, ps2 = 0.f;
  for (int d = tid; d < D_; d += 256) { const float v = s_x[d]; ps += v; ps2 += v * v; }
  ps = wred_sum(ps); ps2 = wred_sum(ps2);
  if (lane == 0) { s_red[wv] = ps; s_red[8 + wv] = ps2; }
  __syncthreads();
  if (tid == 0) {
    float s1 = 0.f, s2 = 0.f;
    for (int i = 0; i < 8; ++i) { s1 += s_red[i]; s2 += s_red[8 + i]; }
    const float mu  = s1 / (float)D_;
    const float var = s2 / (float)D_ - mu * mu;
    s_red[16] = mu;
    s_red[17] = rsqrtf(var + 1e-5f);
  }
  __syncthreads();
  const float mu = s_red[16], rv = s_red[17];
  for (int d = tid; d < D_; d += 256)
    s_x[d] = (s_x[d] - mu) * rv * ln_g[d] + ln_b[d];
  s_x[D_ + tid] = s_qin[tid];             // x = concat(o, src), tid < 256
  __syncthreads();

  // fc1 + relu
  {
    float acc = fc1_b[tid];
    for (int i = 0; i < 1024; ++i) acc += s_x[i] * fc1_w[(size_t)i * 256 + tid];
    s_hdn[tid] = fmaxf(acc, 0.f);
  }
  __syncthreads();
  // fc2
  {
    float acc = fc2_b[tid];
    for (int k = 0; k < 256; ++k) acc += s_hdn[k] * fc2_w[(size_t)k * 256 + tid];
    out[(size_t)b * 256 + tid] = acc;
  }
}

extern "C" void kernel_launch(void* const* d_in, const int* in_sizes, int n_in,
                              void* d_out, int out_size, void* d_ws, size_t ws_size,
                              hipStream_t stream) {
  (void)n_in; (void)out_size; (void)d_ws; (void)ws_size;
  const float* src   = (const float*)d_in[0];
  const float* src_t = (const float*)d_in[1];
  const float* seq   = (const float*)d_in[2];
  const float* seq_t = (const float*)d_in[3];
  const float* seq_e = (const float*)d_in[4];
  const unsigned char* mask = (const unsigned char*)d_in[5];
  const float* Wq    = (const float*)d_in[6];
  const float* Wk    = (const float*)d_in[7];
  const float* Wv    = (const float*)d_in[8];
  const float* Wfc   = (const float*)d_in[9];
  const float* bfc   = (const float*)d_in[10];
  const float* ln_g  = (const float*)d_in[11];
  const float* ln_b  = (const float*)d_in[12];
  const float* fc1_w = (const float*)d_in[13];
  const float* fc1_b = (const float*)d_in[14];
  const float* fc2_w = (const float*)d_in[15];
  const float* fc2_b = (const float*)d_in[16];
  const int B = in_sizes[0] / 256;
  hipLaunchKernelGGL(attn_fused, dim3(B), dim3(256), 0, stream,
                     src, src_t, seq, seq_t, seq_e, mask,
                     Wq, Wk, Wv, Wfc, bfc, ln_g, ln_b,
                     fc1_w, fc1_b, fc2_w, fc2_b, (float*)d_out);
}
